// GroupBlock_55173149884822
// MI455X (gfx1250) — matrix operand order repaired and checked by reading, not compiled, not run on hardware
//
#include <hip/hip_runtime.h>
#include <cstdint>

// ---------------- problem constants ----------------
constexpr int kB  = 8;
constexpr int kC  = 256;
constexpr int kH  = 64;
constexpr int kW  = 160;
constexpr int kHW = kH * kW;            // 10240
constexpr int kNH = 8;
constexpr int kHD = 32;
constexpr int kHID = 1024;
constexpr int kNTOK = kB * kHW;         // 81920
constexpr float kSCALE = 0.17677669529663687f;   // 32^-0.5
constexpr float kEPS = 1e-5f;

// ---------------- WMMA types / helpers ----------------
typedef __attribute__((ext_vector_type(16))) __bf16 v16bf;
typedef __attribute__((ext_vector_type(8)))  float  v8f;

#define DEVINL static __device__ __forceinline__

#if __has_builtin(__builtin_amdgcn_global_load_async_to_lds_b128)
#define HAVE_ASYNC_LDS 1
typedef int vint4_t __attribute__((vector_size(16)));
typedef vint4_t __attribute__((address_space(1)))* gvec4_ptr;   // global (AS1)
typedef vint4_t __attribute__((address_space(3)))* lvec4_ptr;   // LDS    (AS3)
#else
#define HAVE_ASYNC_LDS 0
#endif

DEVINL v8f vzero() {
  v8f z = {0.f, 0.f, 0.f, 0.f, 0.f, 0.f, 0.f, 0.f};
  return z;
}

// cheap f32 -> bf16, round-half-up on the mantissa (2 VALU ops)
DEVINL unsigned short f32_to_bf16(float f) {
  return (unsigned short)((__float_as_uint(f) + 0x8000u) >> 16);
}
DEVINL float bf16_to_f32(unsigned short h) {
  return __uint_as_float(((unsigned int)h) << 16);
}

// packed 2 x (f32 -> bf16) in one 32-bit word (lo first)
DEVINL unsigned int pack_bf16x2(float lo, float hi) {
#if __has_builtin(__builtin_amdgcn_cvt_pk_bf16_f32)
  typedef __bf16 v2bf __attribute__((ext_vector_type(2)));
  union { v2bf v; unsigned int u; } c;
  c.v = __builtin_amdgcn_cvt_pk_bf16_f32(lo, hi);
  return c.u;
#elif __has_builtin(__builtin_amdgcn_perm)
  // v_perm_b32: take high halves of the two rounded words (3 VALU ops total)
  return __builtin_amdgcn_perm(__float_as_uint(hi) + 0x8000u,
                               __float_as_uint(lo) + 0x8000u, 0x07060302u);
#else
  return (unsigned int)f32_to_bf16(lo) | ((unsigned int)f32_to_bf16(hi) << 16);
#endif
}

DEVINL v8f wmma_bf16(v8f acc, v16bf a, v16bf b) {
  return __builtin_amdgcn_wmma_f32_16x16x32_bf16(
      /*neg_a=*/false, a, /*neg_b=*/false, b,
      /*c_mod=*/(short)0, acc, /*reuse_a=*/false, /*reuse_b=*/false);
}

// A fragment: 16x32 bf16, row-major source base[m*ld + k]; per-lane 16 contiguous
// halfs -> backend vectorizes into 2 x b128 loads.
DEVINL v16bf load_a_frag(const unsigned short* base, int ld, int lane) {
  const int m = lane & 15, g = lane >> 4;
  union { v16bf v; unsigned short u[16]; } r;
#pragma unroll
  for (int e = 0; e < 16; ++e) {
    const int j = e >> 1, p = e & 1;
    const int k = (j << 1) + p + (g << 4);
    r.u[e] = base[m * ld + k];
  }
  return r.v;
}

// B fragment: 32x16 bf16 from row-major [N][K]: B[k][n] = base[n*ld + k];
// per-lane k is contiguous -> 2 x b128 loads.
DEVINL v16bf load_b_frag_nk(const unsigned short* base, int ld, int lane) {
  const int n = lane & 15, g = lane >> 4;
  union { v16bf v; unsigned short u[16]; } r;
#pragma unroll
  for (int e = 0; e < 16; ++e) {
    const int j = e >> 1, p = e & 1;
    const int k = (j << 1) + p + (g << 4);
    r.u[e] = base[n * ld + k];
  }
  return r.v;
}

DEVINL float reduce16(float v) {   // reduce across groups of 16 lanes (wave32)
#pragma unroll
  for (int off = 8; off; off >>= 1) v += __shfl_xor(v, off, 16);
  return v;
}

// wait for this wave's outstanding async global->LDS copies
DEVINL void wait_async_copies() {
#if HAVE_ASYNC_LDS
#if __has_builtin(__builtin_amdgcn_s_wait_asynccnt)
  __builtin_amdgcn_s_wait_asynccnt(0);
#else
  asm volatile("s_wait_asynccnt 0" ::: "memory");
#endif
#endif
}

// stage this head's q/k/v weight rows (3 x 32 x 256 bf16 = 48 KB) into an LDS slab.
// Async path: GLOBAL_LOAD_ASYNC_TO_LDS_B128 (ASYNCcnt-tracked, no VGPR round trip).
DEVINL void issue_weight_stage(const unsigned short* __restrict__ Wqkv,
                               unsigned short* __restrict__ sWdst, int h, int tid) {
  for (int i = tid; i < (3 * kHD * kC) / 8; i += 256) {     // 3072 x uint4
    const int rowIdx = i >> 5;                              // 32 uint4 per row
    const int col    = i & 31;
    const int mat = rowIdx >> 5, dd = rowIdx & 31;
    const unsigned short* src = Wqkv + (size_t)(mat * kC + h * kHD + dd) * kC + col * 8;
#if HAVE_ASYNC_LDS
    __builtin_amdgcn_global_load_async_to_lds_b128(
        (gvec4_ptr)const_cast<unsigned short*>(src),
        (lvec4_ptr)(sWdst + (size_t)i * 8),
        /*offset=*/0, /*cpol=*/0);
#else
    ((uint4*)sWdst)[i] = *(const uint4*)src;
#endif
  }
}

// ---------------- kernel 0: f32 -> bf16 weight convert (packed) ----------------
__global__ void cvt_bf16_kernel(const float* __restrict__ in,
                                unsigned int* __restrict__ out, int npairs) {
  int i = blockIdx.x * 256 + threadIdx.x;
  if (i < npairs) {
    const float2 f = ((const float2*)in)[i];
    out[i] = pack_bf16x2(f.x, f.y);
  }
}

// ---------------- kernel 1: [B,C,H,W] -> [B,HW,C] tiled transpose ----------------
__global__ void __launch_bounds__(256)
transpose_chw_hwc_kernel(const float* __restrict__ in, float* __restrict__ out) {
  __shared__ float tile[32][33];
  const int bz  = blockIdx.z;
  const int hw0 = blockIdx.x * 32;
  const int c0  = blockIdx.y * 32;
  const int tx = threadIdx.x, ty = threadIdx.y;   // 32 x 8
#pragma unroll
  for (int j = 0; j < 32; j += 8)
    tile[ty + j][tx] = in[((long long)bz * kC + (c0 + ty + j)) * kHW + hw0 + tx];
  __syncthreads();
#pragma unroll
  for (int j = 0; j < 32; j += 8)
    out[((long long)bz * kHW + hw0 + ty + j) * kC + c0 + tx] = tile[tx][ty + j];
}

// ---------------- axial attention (fused QKV + softmax + PV, in-place residual) ----------------
// one workgroup (8 waves) per sequence; heads processed cooperatively one at a time.
// Per-head weight slabs are double-buffered in LDS; the copy for head h+1 is issued
// (async when available) while head h's S/softmax/PV phases run.
template <int L, bool ALONG_H>
__global__ void __launch_bounds__(256)
attn_kernel(float* __restrict__ X, const unsigned short* __restrict__ Wqkv) {
  constexpr int NT = L / 16;
  constexpr int WSLAB = 3 * kHD * kC;         // 24576 halfs = 48 KB
  extern __shared__ unsigned short smem[];
  unsigned short* sX  = smem;                 // [L][256] bf16
  unsigned short* sQ  = sX + L * kC;          // [L][32]   (pre-scaled by 1/sqrt(d))
  unsigned short* sK  = sQ + L * kHD;         // [L][32]
  unsigned short* sVt = sK + L * kHD;         // [32][L]   (V transposed)
  unsigned short* sP  = sVt + L * kHD;        // [L][L]    (unnormalized exp)
  float*          sInv = (float*)(sP + L * L);// [L]       (1/rowsum)
  unsigned short* sW  = (unsigned short*)(sInv + L);  // 2 x [96][256] slabs

  const int tid  = threadIdx.x;
  const int lane = tid & 31;
  const int wave = tid >> 5;

  const int nseq = ALONG_H ? kW : kH;
  const int b = blockIdx.x / nseq;
  const int s = blockIdx.x % nseq;

  auto rowof = [&](int t) -> long long {
    return ALONG_H ? ((long long)b * kHW + (long long)t * kW + s)
                   : ((long long)b * kHW + (long long)s * kW + t);
  };

  // prologue: kick off head-0 weight staging, then stage the sequence tokens
  issue_weight_stage(Wqkv, sW, 0, tid);
  {
    unsigned int* sX32 = (unsigned int*)sX;
    for (int i = tid; i < L * (kC / 2); i += 256) {
      const int t = i >> 7, c2 = i & 127;
      const float2 f = ((const float2*)X)[rowof(t) * (kC / 2) + c2];
      sX32[t * (kC / 2) + c2] = pack_bf16x2(f.x, f.y);
    }
  }

  for (int h = 0; h < kNH; ++h) {
    unsigned short* sWcur = sW + (h & 1) * WSLAB;
    unsigned short* sWnxt = sW + ((h + 1) & 1) * WSLAB;

    // current slab (and, for h==0, sX) must be resident in LDS for all waves
    wait_async_copies();
    __syncthreads();

    // ---- stage 1: Q/K/V = Xseq @ Wslab^T  (tiles: 3 mats x NT x 2), LDS-fed,
    //      software-pipelined so each k-step's loads overlap the previous WMMA ----
    const int qkvTiles = 3 * NT * 2;
    for (int tIdx = wave; tIdx < qkvTiles; tIdx += 8) {
      const int mat = tIdx / (NT * 2);
      const int rem = tIdx % (NT * 2);
      const int mt = rem >> 1, nt = rem & 1;
      const unsigned short* abase = sX + mt * 16 * kC;
      const unsigned short* bbase = sWcur + (mat * kHD + nt * 16) * kC;
      v8f acc = vzero();
      v16bf acur = load_a_frag(abase, kC, lane);
      v16bf bcur = load_b_frag_nk(bbase, kC, lane);
#pragma unroll
      for (int k0 = 0; k0 < kC; k0 += 32) {
        const bool more = (k0 + 32 < kC);
        const v16bf anxt = more ? load_a_frag(abase + k0 + 32, kC, lane) : acur;
        const v16bf bnxt = more ? load_b_frag_nk(bbase + k0 + 32, kC, lane) : bcur;
        acc = wmma_bf16(acc, acur, bcur);
        acur = anxt; bcur = bnxt;
      }
      const int n = lane & 15, g = lane >> 4;
      if (mat == 2) {
        // store transposed: Vt[d][l'] ; per-lane the 8 l' values are contiguous
        union { unsigned int u[4]; uint4 q; } t;
#pragma unroll
        for (int v = 0; v < 4; ++v)
          t.u[v] = pack_bf16x2(acc[2 * v], acc[2 * v + 1]);
        *(uint4*)&sVt[(nt * 16 + n) * L + mt * 16 + (g << 3)] = t.q;
      } else {
        unsigned short* dst = (mat == 0) ? sQ : sK;
        const float scl = (mat == 0) ? kSCALE : 1.f;   // fold 1/sqrt(d) into Q
#pragma unroll
        for (int v = 0; v < 8; ++v) {
          const int m = v + (g << 3);
          dst[(mt * 16 + m) * kHD + nt * 16 + n] = f32_to_bf16(acc[v] * scl);
        }
      }
    }

    // slab consumed by this wave's tiles -> overlap next head's weight fetch
    if (h + 1 < kNH) issue_weight_stage(Wqkv, sWnxt, h + 1, tid);
    __syncthreads();

    // ---- stage 2: S = Q K^T  -> sP (bf16) ----
    for (int tIdx = wave; tIdx < NT * NT; tIdx += 8) {
      const int mt = tIdx / NT, nt = tIdx % NT;
      v8f acc = vzero();
      v16bf a  = load_a_frag(sQ + mt * 16 * kHD, kHD, lane);
      v16bf bb = load_b_frag_nk(sK + nt * 16 * kHD, kHD, lane);  // B[k=d][n=l'] = K[l'][d]
      acc = wmma_bf16(acc, a, bb);
      const int n = lane & 15, g = lane >> 4;
#pragma unroll
      for (int v = 0; v < 8; ++v) {
        const int m = v + (g << 3);
        sP[(mt * 16 + m) * L + nt * 16 + n] = f32_to_bf16(acc[v]);
      }
    }
    __syncthreads();

    // ---- stage 3: row softmax: store unnormalized exp + 1/rowsum ----
    for (int r = tid; r < L; r += 256) {
      unsigned int* row = (unsigned int*)(sP + r * L);
      float mx = -3.4e38f;
      for (int j = 0; j < L / 2; ++j) {
        const unsigned int w2 = row[j];
        mx = fmaxf(mx, fmaxf(bf16_to_f32((unsigned short)w2),
                             bf16_to_f32((unsigned short)(w2 >> 16))));
      }
      float sum = 0.f;
      for (int j = 0; j < L / 2; ++j) {
        const unsigned int w2 = row[j];
        const float e0 = __expf(bf16_to_f32((unsigned short)w2) - mx);
        const float e1 = __expf(bf16_to_f32((unsigned short)(w2 >> 16)) - mx);
        sum += e0 + e1;
        row[j] = pack_bf16x2(e0, e1);
      }
      sInv[r] = 1.f / sum;       // applied in the PV epilogue
    }
    __syncthreads();

    // ---- stage 4: O = softmax(S) @ V, residual-add into X (exclusive ownership) ----
    for (int tIdx = wave; tIdx < NT * 2; tIdx += 8) {
      const int mt = tIdx >> 1, nt = tIdx & 1;
      const unsigned short* abase = sP + mt * 16 * L;
      const unsigned short* bbase = sVt + (nt * 16) * L;   // B[k=l'][n=d] = Vt[d][l']
      v8f acc = vzero();
      v16bf acur = load_a_frag(abase, L, lane);
      v16bf bcur = load_b_frag_nk(bbase, L, lane);
#pragma unroll
      for (int k0 = 0; k0 < L; k0 += 32) {
        const bool more = (k0 + 32 < L);
        const v16bf anxt = more ? load_a_frag(abase + k0 + 32, L, lane) : acur;
        const v16bf bnxt = more ? load_b_frag_nk(bbase + k0 + 32, L, lane) : bcur;
        acc = wmma_bf16(acc, acur, bcur);
        acur = anxt; bcur = bnxt;
      }
      const int n = lane & 15, g = lane >> 4;
#pragma unroll
      for (int v = 0; v < 8; ++v) {
        const int m = v + (g << 3);
        const int t = mt * 16 + m;
        const int c = h * kHD + nt * 16 + n;
        X[rowof(t) * kC + c] += acc[v] * sInv[t];   // contiguous per-lane sInv reads
      }
    }
    __syncthreads();   // sQ/sK/sVt/sP/sInv reused next head
  }
}

// ---------------- fused LN1 -> fc1+GELU -> fc2 -> residual -> LN2 megakernel ----------------
// 16 tokens per block, 8 waves; fc1 activation [16][1024] lives only in LDS.
// Weight (B) fragments are software double-buffered so global-load latency
// overlaps the WMMA instead of draining before each one.
template <bool TRANS_OUT>
__global__ void __launch_bounds__(256)
mlp_kernel(const float* __restrict__ Xin,
           const unsigned short* __restrict__ W1, const float* __restrict__ Bi1,
           const unsigned short* __restrict__ W2, const float* __restrict__ Bi2,
           const float* __restrict__ g1, const float* __restrict__ b1,
           const float* __restrict__ g2, const float* __restrict__ b2,
           float* __restrict__ Out) {
  __shared__ unsigned short sXln[16 * kC];     // 8 KB  (post-LN1, also residual)
  __shared__ unsigned short sHmid[16 * kHID];  // 32 KB (GELU(fc1))
  __shared__ float          sY[16 * kC];       // 16 KB

  const int tid  = threadIdx.x;
  const int lane = tid & 31;
  const int wave = tid >> 5;
  const long long t0 = (long long)blockIdx.x * 16;
  const int r  = tid >> 4;     // 0..15 : token row
  const int lg = tid & 15;     // 16 lanes per row, each owns 16 contiguous channels
  const int cb = lg << 4;      // channel base for this lane

  // ---- LN1 (blocked channels -> vectorizable global reads / LDS writes) ----
  float xv[16];
  float sum = 0.f;
#pragma unroll
  for (int j = 0; j < 16; ++j) {
    xv[j] = Xin[(t0 + r) * kC + cb + j];
    sum += xv[j];
  }
  sum = reduce16(sum);
  const float mean = sum * (1.f / kC);
  float var = 0.f;
#pragma unroll
  for (int j = 0; j < 16; ++j) { const float d = xv[j] - mean; var += d * d; }
  var = reduce16(var);
  const float rs = rsqrtf(var * (1.f / kC) + kEPS);
  {
    unsigned int packed[8];
#pragma unroll
    for (int j = 0; j < 8; ++j) {
      const float v0 = (xv[2 * j]     - mean) * rs * g1[cb + 2 * j]     + b1[cb + 2 * j];
      const float v1 = (xv[2 * j + 1] - mean) * rs * g1[cb + 2 * j + 1] + b1[cb + 2 * j + 1];
      packed[j] = pack_bf16x2(v0, v1);
    }
    uint4* dst = (uint4*)&sXln[r * kC + cb];
    dst[0] = make_uint4(packed[0], packed[1], packed[2], packed[3]);
    dst[1] = make_uint4(packed[4], packed[5], packed[6], packed[7]);
  }
  __syncthreads();

  // ---- fc1 + exact GELU : [16,256] @ [256,1024]^T ----
  for (int nt = wave; nt < kHID / 16; nt += 8) {
    const unsigned short* wrow = W1 + (size_t)(nt * 16) * kC;
    v8f acc = vzero();
    v16bf bcur = load_b_frag_nk(wrow, kC, lane);
#pragma unroll
    for (int k0 = 0; k0 < kC; k0 += 32) {
      const v16bf bnxt = (k0 + 32 < kC) ? load_b_frag_nk(wrow + k0 + 32, kC, lane) : bcur;
      const v16bf a = load_a_frag(sXln + k0, kC, lane);
      acc = wmma_bf16(acc, a, bcur);
      bcur = bnxt;
    }
    const int n = lane & 15, g = lane >> 4;
    const int d = nt * 16 + n;
    const float bias = Bi1[d];
#pragma unroll
    for (int v = 0; v < 8; ++v) {
      const int m = v + (g << 3);
      float val = acc[v] + bias;
      val = 0.5f * val * (1.f + erff(val * 0.70710678118654752f));
      sHmid[m * kHID + d] = f32_to_bf16(val);
    }
  }
  __syncthreads();

  // ---- fc2 + bias + residual : [16,1024] @ [1024,256]^T ----
  for (int nt = wave; nt < kC / 16; nt += 8) {
    const unsigned short* wrow = W2 + (size_t)(nt * 16) * kHID;
    v8f acc = vzero();
    v16bf bcur = load_b_frag_nk(wrow, kHID, lane);
#pragma unroll
    for (int k0 = 0; k0 < kHID; k0 += 32) {
      const v16bf bnxt = (k0 + 32 < kHID) ? load_b_frag_nk(wrow + k0 + 32, kHID, lane) : bcur;
      const v16bf a = load_a_frag(sHmid + k0, kHID, lane);
      acc = wmma_bf16(acc, a, bcur);
      bcur = bnxt;
    }
    const int n = lane & 15, g = lane >> 4;
    const int c = nt * 16 + n;
    const float bias = Bi2[c];
#pragma unroll
    for (int v = 0; v < 8; ++v) {
      const int m = v + (g << 3);
      sY[m * kC + c] = acc[v] + bias + bf16_to_f32(sXln[m * kC + c]);
    }
  }
  __syncthreads();

  // ---- LN2 + write ----
  float yv[16];
  float s2 = 0.f;
#pragma unroll
  for (int j = 0; j < 16; ++j) { yv[j] = sY[r * kC + cb + j]; s2 += yv[j]; }
  s2 = reduce16(s2);
  const float mean2 = s2 * (1.f / kC);
  float var2 = 0.f;
#pragma unroll
  for (int j = 0; j < 16; ++j) { const float d = yv[j] - mean2; var2 += d * d; }
  var2 = reduce16(var2);
  const float rs2 = rsqrtf(var2 * (1.f / kC) + kEPS);
#pragma unroll
  for (int j = 0; j < 16; ++j) {
    const int c = cb + j;
    const float val = (yv[j] - mean2) * rs2 * g2[c] + b2[c];
    if (TRANS_OUT) {
      const long long t = t0 + r;
      const int w   = (int)(t % kW);
      const int hh  = (int)((t / kW) % kH);
      const int bb2 = (int)(t / ((long long)kW * kH));
      Out[(((long long)bb2 * kC + c) * kH + hh) * kW + w] = val;
    } else {
      Out[(t0 + r) * kC + c] = val;
    }
  }
  (void)lane; (void)wave;
}

// ---------------- host orchestration ----------------
extern "C" void kernel_launch(void* const* d_in, const int* in_sizes, int n_in,
                              void* d_out, int out_size, void* d_ws, size_t ws_size,
                              hipStream_t stream) {
  (void)in_sizes; (void)n_in; (void)out_size; (void)ws_size;

  const float* x      = (const float*)d_in[0];
  const float* Wqkv_h = (const float*)d_in[1];
  const float* Wqkv_w = (const float*)d_in[2];
  const float* fc1_w1 = (const float*)d_in[3];
  const float* fc1_b1 = (const float*)d_in[4];
  const float* fc2_w1 = (const float*)d_in[5];
  const float* fc2_b1 = (const float*)d_in[6];
  const float* fc1_w2 = (const float*)d_in[7];
  const float* fc1_b2 = (const float*)d_in[8];
  const float* fc2_w2 = (const float*)d_in[9];
  const float* fc2_b2 = (const float*)d_in[10];
  const float* g11 = (const float*)d_in[11]; const float* b11 = (const float*)d_in[12];
  const float* g12 = (const float*)d_in[13]; const float* b12 = (const float*)d_in[14];
  const float* g21 = (const float*)d_in[15]; const float* b21 = (const float*)d_in[16];
  const float* g22 = (const float*)d_in[17]; const float* b22 = (const float*)d_in[18];

  // workspace carve
  const size_t XC_BYTES = (size_t)kNTOK * kC * sizeof(float);    // 83.9 MB channel-last tensor
  float* Xc = (float*)d_ws;
  unsigned short* wb = (unsigned short*)((char*)d_ws + XC_BYTES);
  const int WQKV_E = 3 * kC * kC;        // 196608
  const int FC1_E  = kHID * kC;          // 262144
  const int FC2_E  = kC * kHID;          // 262144
  unsigned short* wqkvh_bf = wb;
  unsigned short* wqkvw_bf = wqkvh_bf + WQKV_E;
  unsigned short* fc1w1_bf = wqkvw_bf + WQKV_E;
  unsigned short* fc2w1_bf = fc1w1_bf + FC1_E;
  unsigned short* fc1w2_bf = fc2w1_bf + FC2_E;
  unsigned short* fc2w2_bf = fc1w2_bf + FC1_E;

  // 0) weights -> bf16 (packed pair per thread)
  cvt_bf16_kernel<<<(WQKV_E / 2 + 255) / 256, 256, 0, stream>>>(Wqkv_h, (unsigned int*)wqkvh_bf, WQKV_E / 2);
  cvt_bf16_kernel<<<(WQKV_E / 2 + 255) / 256, 256, 0, stream>>>(Wqkv_w, (unsigned int*)wqkvw_bf, WQKV_E / 2);
  cvt_bf16_kernel<<<(FC1_E / 2 + 255) / 256, 256, 0, stream>>>(fc1_w1, (unsigned int*)fc1w1_bf, FC1_E / 2);
  cvt_bf16_kernel<<<(FC2_E / 2 + 255) / 256, 256, 0, stream>>>(fc2_w1, (unsigned int*)fc2w1_bf, FC2_E / 2);
  cvt_bf16_kernel<<<(FC1_E / 2 + 255) / 256, 256, 0, stream>>>(fc1_w2, (unsigned int*)fc1w2_bf, FC1_E / 2);
  cvt_bf16_kernel<<<(FC2_E / 2 + 255) / 256, 256, 0, stream>>>(fc2_w2, (unsigned int*)fc2w2_bf, FC2_E / 2);

  // 1) [B,C,H,W] -> [B,HW,C]
  transpose_chw_hwc_kernel<<<dim3(kHW / 32, kC / 32, kB), dim3(32, 8), 0, stream>>>(x, Xc);

  const int SMEM_WGT = 2 * 3 * kHD * kC * (int)sizeof(unsigned short);   // 2 x 48 KB slabs

  // 2) H-axis attention (L=64), in-place residual  (~148 KB LDS)
  const int SMEM_H = (kH * kC + 3 * kH * kHD + kH * kH) * (int)sizeof(unsigned short)
                   + kH * (int)sizeof(float) + SMEM_WGT;
  (void)hipFuncSetAttribute(reinterpret_cast<const void*>(&attn_kernel<kH, true>),
                            hipFuncAttributeMaxDynamicSharedMemorySize, SMEM_H);
  attn_kernel<kH, true><<<kB * kW, 256, SMEM_H, stream>>>(Xc, wqkvh_bf);

  // 3) fused LN11 -> MLP1 -> LN12 (in place on Xc)
  mlp_kernel<false><<<kNTOK / 16, 256, 0, stream>>>(Xc, fc1w1_bf, fc1_b1, fc2w1_bf, fc2_b1,
                                                    g11, b11, g12, b12, Xc);

  // 4) W-axis attention (L=160), in-place residual  (~257 KB LDS of the WGP's 320 KB)
  const int SMEM_W = (kW * kC + 3 * kW * kHD + kW * kW) * (int)sizeof(unsigned short)
                   + kW * (int)sizeof(float) + SMEM_WGT;
  (void)hipFuncSetAttribute(reinterpret_cast<const void*>(&attn_kernel<kW, false>),
                            hipFuncAttributeMaxDynamicSharedMemorySize, SMEM_W);
  attn_kernel<kW, false><<<kB * kH, 256, SMEM_W, stream>>>(Xc, wqkvw_bf);

  // 5) fused LN21 -> MLP2 -> LN22, writing transposed [B,C,H,W] straight to d_out
  mlp_kernel<true><<<kNTOK / 16, 256, 0, stream>>>(Xc, fc1w2_bf, fc1_b2, fc2w2_bf, fc2_b2,
                                                   g21, b21, g22, b22, (float*)d_out);
}
